// MTS3_3_level_87926570484129
// MI455X (gfx1250) — compile-verified, matrix operand order checked by reading, not run on hardware
//
#include <hip/hip_runtime.h>
#include <math.h>

typedef _Float16 f16;
typedef __attribute__((ext_vector_type(16))) _Float16 v16h;
typedef __attribute__((ext_vector_type(8)))  float    v8f;

#define ROWS 153600   // B*T
#define Bsz  256
#define Tlen 600
#define LODc 30
#define LSDc 60
#define Kc   16

// 16-bit A/B fragment K mapping: fragment element j (0..15) of lane-half h
// holds K = base(vgpr) + h*8 + pos   (per CDNA5 ISA 16-bit A layout).
__device__ __forceinline__ int kmap16(int j, int half) {
  int vg = j >> 1, pos = j & 1;
  int base = (vg < 4) ? (vg * 2) : (16 + (vg - 4) * 2);
  return base + half * 8 + pos;
}

// Inverse of kmap16 within a 32-wide K slab: kk (0..31) -> (j, halfk)
__device__ __forceinline__ void kinv16(int kk, int& j, int& halfk) {
  halfk = (kk >> 3) & 1;
  int kkp = kk - halfk * 8;  // in {0..7} U {16..23}
  int pos = kkp & 1;
  int vg = (kkp < 8) ? (kkp >> 1) : (4 + ((kkp - 16) >> 1));
  j = vg * 2 + pos;
}

__device__ __forceinline__ float softplusf(float x) {
  return x > 20.f ? x : log1pf(expf(x));
}

// ---------------------------------------------------------------------------
// Swizzle W1 (src [sr][256] f32) into fragment order:
// dst[slab][tile(16)][lane(32)][j(16)] f16, slab = K/32, zero-padded rows.
__global__ void conv_w1_swz_kernel(const float* __restrict__ src, int sr,
                                   f16* __restrict__ dst, int kpad) {
  int i = blockIdx.x * 256 + threadIdx.x;
  int total = kpad * 256;
  if (i >= total) return;
  int j    = i & 15;
  int lane = (i >> 4) & 31;
  int tile = (i >> 9) & 15;
  int slab = i >> 13;
  int k = slab * 32 + kmap16(j, lane >> 4);
  int n = tile * 16 + (lane & 15);
  float v = (k < sr) ? src[(size_t)k * 256 + n] : 0.f;
  dst[i] = (f16)v;
}

// Swizzle head weights (src [256][sc] f32) into fragment order:
// dst[slab(8)][tile(4)][lane(32)][j(16)] f16, cols zero-padded to 64.
__global__ void conv_head_swz_kernel(const float* __restrict__ src, int sc,
                                     f16* __restrict__ dst) {
  int i = blockIdx.x * 256 + threadIdx.x;
  if (i >= 16384) return;
  int j    = i & 15;
  int lane = (i >> 4) & 31;
  int tile = (i >> 9) & 3;
  int slab = i >> 11;
  int k = slab * 32 + kmap16(j, lane >> 4);
  int c = tile * 16 + (lane & 15);
  float v = (c < sc) ? src[(size_t)k * sc + c] : 0.f;
  dst[i] = (f16)v;
}

// ---------------------------------------------------------------------------
// Fused encoder / MLP:  H = relu(X@W1+b1);  mean = H@Wm+bm;  var = softplus(H@Wv+bv)+1e-4
// All WMMA operands stored fragment-swizzled so each fragment is one v16h
// (32B contiguous) LDS load -> 2x ds_load_b128 per fragment.
__global__ void __launch_bounds__(256)
enc_wmma_kernel(const float* __restrict__ X, int dinRaw, int tembPeriod,
                const f16* __restrict__ W1s, int kpad,
                const float* __restrict__ b1,
                const f16* __restrict__ Wms, const float* __restrict__ bm,
                const f16* __restrict__ Wvs, const float* __restrict__ bv,
                float* __restrict__ outMean, float* __restrict__ outVar,
                int outStride, int dout, int rows, int spMean) {
  __shared__ __align__(32) f16 Xs[4 * 32 * 16];       // 4KB  A slab (swizzled)
  __shared__ __align__(32) f16 Ws[16 * 32 * 16];      // 16KB W1 slab (swizzled)
  __shared__ __align__(32) f16 Hs[8 * 4 * 32 * 16];   // 32KB hidden (swizzled)
  __shared__ __align__(32) f16 HW[2 * 4 * 32 * 16];   // 8KB  head slabs (swizzled)

  const int tid  = threadIdx.x;
  const int wave = tid >> 5, lane = tid & 31;
  const int half = lane >> 4, lm = lane & 15;
  const int rowBase = blockIdx.x * 64;
  const int strip = wave & 3;      // 16-row strip
  const int chalf = wave >> 2;     // 128-col half of hidden

  v8f acc[8] = {};

  for (int kb = 0; kb < kpad; kb += 32) {
    const int slab = kb >> 5;
    // stage A slab directly in fragment order (f32->f16 + temb + pad)
    for (int i = tid; i < 2048; i += 256) {
      int j  = i & 15;
      int ln2 = (i >> 4) & 31;
      int st = i >> 9;
      int r  = st * 16 + (ln2 & 15);
      int k  = kb + kmap16(j, ln2 >> 4);
      int gr = rowBase + r;
      float v = 0.f;
      if (gr < rows) {
        if (k < dinRaw) v = X[(size_t)gr * dinRaw + k];
        else if (tembPeriod > 0 && k == dinRaw) {
          int t = gr % Tlen;
          v = (float)(t % tembPeriod) / (float)tembPeriod;
        }
      }
      Xs[i] = (f16)v;
    }
    // stage W1 slab: straight vector copy (already swizzled in global)
    {
      const uint4* s = (const uint4*)(W1s + (size_t)slab * 8192);
      uint4* d = (uint4*)Ws;
      for (int i = tid; i < 1024; i += 256) d[i] = s[i];
    }
    __syncthreads();

    v16h af = *(const v16h*)&Xs[(strip * 32 + lane) * 16];
#pragma unroll
    for (int ct = 0; ct < 8; ++ct) {
      v16h bf = *(const v16h*)&Ws[((chalf * 8 + ct) * 32 + lane) * 16];
      acc[ct] = __builtin_amdgcn_wmma_f32_16x16x32_f16(false, af, false, bf,
                                                       (short)0, acc[ct], false, false);
    }
    __syncthreads();
  }

  // H = relu(acc + b1) -> f16 in LDS, stored fragment-swizzled for stage 2
#pragma unroll
  for (int ct = 0; ct < 8; ++ct) {
    int n0 = (chalf * 8 + ct) * 16;
#pragma unroll
    for (int v = 0; v < 8; ++v) {
      int row = strip * 16 + v + half * 8;
      int col = n0 + lm;
      float x = acc[ct][v] + b1[col];
      x = x > 0.f ? x : 0.f;
      int slab = col >> 5, kk = col & 31;
      int j, hk;
      kinv16(kk, j, hk);
      int lane2 = hk * 16 + (row & 15);
      Hs[((slab * 4 + strip) * 32 + lane2) * 16 + j] = (f16)x;
    }
  }
  __syncthreads();

  // stage 2: waves 0-3 -> mean head, waves 4-7 -> var head (dummy if absent)
  const int head = wave >> 2;
  v8f acc2[4] = {};

  for (int kb = 0; kb < 256; kb += 32) {
    const int slab = kb >> 5;
    {
      const uint4* sm = (const uint4*)(Wms + (size_t)slab * 2048);
      const uint4* sv = (const uint4*)((Wvs ? Wvs : Wms) + (size_t)slab * 2048);
      uint4* d = (uint4*)HW;
      for (int i = tid; i < 512; i += 256) d[i] = (i < 256) ? sm[i] : sv[i - 256];
    }
    __syncthreads();
    v16h af = *(const v16h*)&Hs[((slab * 4 + strip) * 32 + lane) * 16];
#pragma unroll
    for (int ct = 0; ct < 4; ++ct) {
      v16h bf = *(const v16h*)&HW[((head * 4 + ct) * 32 + lane) * 16];
      acc2[ct] = __builtin_amdgcn_wmma_f32_16x16x32_f16(false, af, false, bf,
                                                        (short)0, acc2[ct], false, false);
    }
    __syncthreads();
  }

  if (head == 0) {
#pragma unroll
    for (int ct = 0; ct < 4; ++ct) {
      int c = ct * 16 + lm;
      if (c < dout) {
#pragma unroll
        for (int v = 0; v < 8; ++v) {
          int gr = rowBase + strip * 16 + v + half * 8;
          if (gr < rows) {
            float x = acc2[ct][v] + bm[c];
            if (spMean) x = softplusf(x) + 1e-4f;
            outMean[(size_t)gr * outStride + c] = x;
          }
        }
      }
    }
  } else if (Wvs) {
#pragma unroll
    for (int ct = 0; ct < 4; ++ct) {
      int c = ct * 16 + lm;
      if (c < dout) {
#pragma unroll
        for (int v = 0; v < 8; ++v) {
          int gr = rowBase + strip * 16 + v + half * 8;
          if (gr < rows) {
            float x = softplusf(acc2[ct][v] + bv[c]) + 1e-4f;
            outVar[(size_t)gr * outStride + c] = x;
          }
        }
      }
    }
  }
}

// ---------------------------------------------------------------------------
// _update_set reductions: S1 = sum(valid/var), S2 = sum(valid*w/var) per chunk.
__global__ void reduce_obs_kernel(const float* __restrict__ encM, const float* __restrict__ encV,
                                  const unsigned char* __restrict__ valid,
                                  int chunkLen, int nch,
                                  float* __restrict__ S1, float* __restrict__ S2) {
  int idx = blockIdx.x * 256 + threadIdx.x;
  int total = Bsz * nch * LODc;
  if (idx >= total) return;
  int d = idx % LODc;
  int c = (idx / LODc) % nch;
  int b = idx / (LODc * nch);
  float p = 0.f, q = 0.f;
  for (int i = 0; i < chunkLen; ++i) {
    size_t r = (size_t)b * Tlen + c * chunkLen + i;
    float var = encV[r * 64 + d];
    float w   = encM[r * 64 + d];
    float v   = valid[r] ? 1.f : 0.f;
    p += v / var;
    q += v * w / var;
  }
  S1[idx] = p;
  S2[idx] = q;
}

// _infer_act reductions: SA = sum(1/av), SB = sum(am/av) per chunk.
__global__ void reduce_act_kernel(const float* __restrict__ encM, const float* __restrict__ encV,
                                  int chunkLen, int nch,
                                  float* __restrict__ SA, float* __restrict__ SB) {
  int idx = blockIdx.x * 256 + threadIdx.x;
  int total = Bsz * nch * LSDc;
  if (idx >= total) return;
  int d = idx % LSDc;
  int c = (idx / LSDc) % nch;
  int b = idx / (LSDc * nch);
  float sa = 0.f, sb = 0.f;
  for (int i = 0; i < chunkLen; ++i) {
    size_t r = (size_t)b * Tlen + c * chunkLen + i;
    float av = encV[r * 64 + d];
    sa += 1.f / av;
    sb += encM[r * 64 + d] / av;
  }
  SA[idx] = sa;
  SB[idx] = sb;
}

// ---------------------------------------------------------------------------
// Level 1/2 recurrence: update_set + predict, one block per batch element.
__global__ void __launch_bounds__(64)
level_rec_kernel(const float* __restrict__ S1, const float* __restrict__ S2,
                 const float* __restrict__ SA, const float* __restrict__ SB,
                 const float* __restrict__ wc, const float* __restrict__ bc,
                 const float* __restrict__ tm, const float* __restrict__ ln,
                 float* __restrict__ postOut, float* __restrict__ priorOut,
                 float* __restrict__ absOut,
                 int nch, int predMod, int nprior) {
  const int b = blockIdx.x, tid = threadIdx.x;
  __shared__ float wcS[LSDc * Kc], bcS[Kc], tmS[4 * Kc * LODc], elnS[LSDc];
  __shared__ float mu[LODc], ml[LODc], cu[LODc], cl[LODc], cs[LODc];
  __shared__ float post[LSDc], pcu[LODc], pcl[LODc], pcs[LODc];
  __shared__ float logits[Kc], al[Kc], A[4 * LODc], aam[LSDc], aav[LSDc];

  for (int i = tid; i < LSDc * Kc; i += 64) wcS[i] = wc[i];
  for (int i = tid; i < 4 * Kc * LODc; i += 64) tmS[i] = tm[i];
  if (tid < Kc) bcS[tid] = bc[tid];
  for (int i = tid; i < LSDc; i += 64) elnS[i] = expf(ln[i]);
  for (int i = tid; i < LODc; i += 64) {
    mu[i] = 0.f; ml[i] = 0.f; cu[i] = 10.f; cl[i] = 10.f; cs[i] = 1.f;  // ISC=10
  }
  __syncthreads();

  int pi = 0;
  for (int c = 0; c < nch; ++c) {
    if (tid < LODc) {
      int d = tid;
      size_t o = ((size_t)b * nch + c) * LODc + d;
      float prec = S1[o];
      float ve = 1.f / fmaxf(prec, 1e-8f);
      float we = S2[o] * ve;
      float denom = cu[d] + ve;
      float qu = cu[d] / denom, ql = cs[d] / denom;
      float rr = we - mu[d];
      float pm = mu[d] + qu * rr, pl = ml[d] + ql * rr;
      pcu[d] = (1.f - qu) * cu[d];
      pcl[d] = cl[d] - ql * cs[d];
      pcs[d] = (1.f - qu) * cs[d];
      post[d] = pm; post[LODc + d] = pl;
      size_t po = ((size_t)b * nch + c) * LSDc;
      postOut[po + d] = pm;
      postOut[po + LODc + d] = pl;
    }
    __syncthreads();

    if ((c % predMod) != predMod - 1) {
      for (int i = tid; i < LSDc; i += 64) {
        size_t o = ((size_t)b * nch + c + 1) * LSDc + i;
        float av = 1.f / (1.f + SA[o]);
        aav[i] = av;
        aam[i] = av * SB[o];
        if (absOut) absOut[((size_t)b * nprior + pi) * LSDc + i] = aam[i];
      }
      __syncthreads();
      if (tid < Kc) {
        float s = bcS[tid];
        for (int j = 0; j < LSDc; ++j) s += post[j] * wcS[j * Kc + tid];
        logits[tid] = s;
      }
      __syncthreads();
      if (tid == 0) {
        float m = logits[0];
        for (int k = 1; k < Kc; ++k) m = fmaxf(m, logits[k]);
        float s = 0.f;
        for (int k = 0; k < Kc; ++k) { float e = expf(logits[k] - m); al[k] = e; s += e; }
        float inv = 1.f / s;
        for (int k = 0; k < Kc; ++k) al[k] *= inv;
      }
      __syncthreads();
      for (int idx = tid; idx < 4 * LODc; idx += 64) {
        int i = idx / LODc, d = idx - i * LODc;
        float s = 0.f;
        for (int k = 0; k < Kc; ++k) s += al[k] * tmS[(i * Kc + k) * LODc + d];
        A[idx] = s;
      }
      __syncthreads();
      if (tid < LODc) {
        int d = tid;
        float A11 = A[d], A12 = A[LODc + d], A21 = A[2 * LODc + d], A22 = A[3 * LODc + d];
        float bu = aam[d], bl = aam[LODc + d];
        float vu = aav[d], vl = aav[LODc + d];
        float nu = elnS[d], nl = elnS[LODc + d];
        float pm = post[d], pl = post[LODc + d];
        float nmu = A11 * pm + A12 * pl + bu;
        float nml = A21 * pm + A22 * pl + bl;
        float ncu = A11 * A11 * pcu[d] + 2.f * A11 * A12 * pcs[d] + A12 * A12 * pcl[d] + vu + nu;
        float ncl = A21 * A21 * pcu[d] + 2.f * A21 * A22 * pcs[d] + A22 * A22 * pcl[d] + vl + nl;
        float ncs = A11 * A21 * pcu[d] + (A11 * A22 + A12 * A21) * pcs[d] + A12 * A22 * pcl[d];
        mu[d] = nmu; ml[d] = nml; cu[d] = ncu; cl[d] = ncl; cs[d] = ncs;
        size_t po = ((size_t)b * nprior + pi) * LSDc;
        priorOut[po + d] = nmu;
        priorOut[po + LODc + d] = nml;
      }
      ++pi;
      __syncthreads();
    }
  }
}

// ---------------------------------------------------------------------------
// b_seq = actions @ wctrl + repeat(post_task_means, H) @ wtask
__global__ void bseq_kernel(const float* __restrict__ act, const float* __restrict__ ptm,
                            const float* __restrict__ wctrl, const float* __restrict__ wtask,
                            float* __restrict__ bseq) {
  int idx = blockIdx.x * 256 + threadIdx.x;
  if (idx >= ROWS * LSDc) return;
  int r = idx / LSDc, d = idx - r * LSDc;
  int b = r / Tlen, t = r - b * Tlen;
  float s = 0.f;
  const float* a = act + (size_t)r * 8;
  for (int j = 0; j < 8; ++j) s += a[j] * wctrl[j * LSDc + d];
  const float* task = ptm + ((size_t)b * 40 + t / 15) * LSDc;
  for (int j = 0; j < LSDc; ++j) s += task[j] * wtask[j * LSDc + d];
  bseq[idx] = s;
}

// ---------------------------------------------------------------------------
// Level-3 600-step scan: update_one + predict (b_var = 0), one block per batch.
__global__ void __launch_bounds__(64)
scan_kernel(const float* __restrict__ encM, const float* __restrict__ encV,
            const unsigned char* __restrict__ valid, const float* __restrict__ bseq,
            const float* __restrict__ wc, const float* __restrict__ bc,
            const float* __restrict__ tm, const float* __restrict__ ln,
            float* __restrict__ psm, float* __restrict__ psc) {
  const int b = blockIdx.x, tid = threadIdx.x;
  __shared__ float wcS[LSDc * Kc], bcS[Kc], tmS[4 * Kc * LODc], elnS[LSDc];
  __shared__ float mu[LODc], ml[LODc], cu[LODc], cl[LODc], cs[LODc];
  __shared__ float post[LSDc], pcu[LODc], pcl[LODc], pcs[LODc];
  __shared__ float logits[Kc], al[Kc], A[4 * LODc];

  for (int i = tid; i < LSDc * Kc; i += 64) wcS[i] = wc[i];
  for (int i = tid; i < 4 * Kc * LODc; i += 64) tmS[i] = tm[i];
  if (tid < Kc) bcS[tid] = bc[tid];
  for (int i = tid; i < LSDc; i += 64) elnS[i] = expf(ln[i]);
  for (int i = tid; i < LODc; i += 64) {
    mu[i] = 0.f; ml[i] = 0.f; cu[i] = 1.f; cl[i] = 1.f; cs[i] = 1.f;
  }
  __syncthreads();

  for (int t = 0; t < Tlen; ++t) {
    size_t r = (size_t)b * Tlen + t;
    float g = valid[r] ? 1.f : 0.f;
    if (tid < LODc) {
      int d = tid;
      float var = encV[r * 64 + d];
      float w   = encM[r * 64 + d];
      float denom = cu[d] + var;
      float qu = g * cu[d] / denom, ql = g * cs[d] / denom;
      float rr = w - mu[d];
      float pm = mu[d] + qu * rr, pl = ml[d] + ql * rr;
      float u = (1.f - qu) * cu[d];
      float l = cl[d] - ql * cs[d];
      float sxy = (1.f - qu) * cs[d];
      post[d] = pm; post[LODc + d] = pl;
      pcu[d] = u; pcl[d] = l; pcs[d] = sxy;
      psm[r * LSDc + d] = pm;
      psm[r * LSDc + LODc + d] = pl;
      psc[r * 90 + d] = u;
      psc[r * 90 + LODc + d] = l;
      psc[r * 90 + 2 * LODc + d] = sxy;
    }
    __syncthreads();
    if (tid < Kc) {
      float s = bcS[tid];
      for (int j = 0; j < LSDc; ++j) s += post[j] * wcS[j * Kc + tid];
      logits[tid] = s;
    }
    __syncthreads();
    if (tid == 0) {
      float m = logits[0];
      for (int k = 1; k < Kc; ++k) m = fmaxf(m, logits[k]);
      float s = 0.f;
      for (int k = 0; k < Kc; ++k) { float e = expf(logits[k] - m); al[k] = e; s += e; }
      float inv = 1.f / s;
      for (int k = 0; k < Kc; ++k) al[k] *= inv;
    }
    __syncthreads();
    for (int idx = tid; idx < 4 * LODc; idx += 64) {
      int i = idx / LODc, d = idx - i * LODc;
      float s = 0.f;
      for (int k = 0; k < Kc; ++k) s += al[k] * tmS[(i * Kc + k) * LODc + d];
      A[idx] = s;
    }
    __syncthreads();
    if (tid < LODc) {
      int d = tid;
      float A11 = A[d], A12 = A[LODc + d], A21 = A[2 * LODc + d], A22 = A[3 * LODc + d];
      float bu = bseq[r * LSDc + d], bl = bseq[r * LSDc + LODc + d];
      float nu = elnS[d], nl = elnS[LODc + d];
      float pm = post[d], pl = post[LODc + d];
      mu[d] = A11 * pm + A12 * pl + bu;
      ml[d] = A21 * pm + A22 * pl + bl;
      cu[d] = A11 * A11 * pcu[d] + 2.f * A11 * A12 * pcs[d] + A12 * A12 * pcl[d] + nu;
      cl[d] = A21 * A21 * pcu[d] + 2.f * A21 * A22 * pcs[d] + A22 * A22 * pcl[d] + nl;
      cs[d] = A11 * A21 * pcu[d] + (A11 * A22 + A12 * A21) * pcs[d] + A12 * A22 * pcl[d];
    }
    __syncthreads();
  }
}

// ---------------------------------------------------------------------------
extern "C" void kernel_launch(void* const* d_in, const int* in_sizes, int n_in,
                              void* d_out, int out_size, void* d_ws, size_t ws_size,
                              hipStream_t stream) {
  (void)in_sizes; (void)n_in; (void)out_size; (void)ws_size;

  const float* obs = (const float*)d_in[0];
  const float* act = (const float*)d_in[1];
  const unsigned char* valid = (const unsigned char*)d_in[2];
  const float* aoe_w1 = (const float*)d_in[3];
  const float* aoe_b1 = (const float*)d_in[4];
  const float* aoe_wm = (const float*)d_in[5];
  const float* aoe_bm = (const float*)d_in[6];
  const float* aoe_wv = (const float*)d_in[7];
  const float* aoe_bv = (const float*)d_in[8];
  const float* ae_w1 = (const float*)d_in[9];
  const float* ae_b1 = (const float*)d_in[10];
  const float* ae_wm = (const float*)d_in[11];
  const float* ae_bm = (const float*)d_in[12];
  const float* ae_wv = (const float*)d_in[13];
  const float* ae_bv = (const float*)d_in[14];
  const float* oe_w1 = (const float*)d_in[15];
  const float* oe_b1 = (const float*)d_in[16];
  const float* oe_wm = (const float*)d_in[17];
  const float* oe_bm = (const float*)d_in[18];
  const float* oe_wv = (const float*)d_in[19];
  const float* oe_bv = (const float*)d_in[20];
  const float* pp_wc = (const float*)d_in[21];
  const float* pp_bc = (const float*)d_in[22];
  const float* pp_tm = (const float*)d_in[23];
  const float* pp_ln = (const float*)d_in[24];
  const float* tp_wc = (const float*)d_in[25];
  const float* tp_bc = (const float*)d_in[26];
  const float* tp_tm = (const float*)d_in[27];
  const float* tp_ln = (const float*)d_in[28];
  const float* sp_wc = (const float*)d_in[29];
  const float* sp_bc = (const float*)d_in[30];
  const float* sp_tm = (const float*)d_in[31];
  const float* sp_ln = (const float*)d_in[32];
  const float* sp_wctrl = (const float*)d_in[33];
  const float* sp_wtask = (const float*)d_in[34];
  const float* dm_w1 = (const float*)d_in[35];
  const float* dm_b1 = (const float*)d_in[36];
  const float* dm_w2 = (const float*)d_in[37];
  const float* dm_b2 = (const float*)d_in[38];
  const float* dv_w1 = (const float*)d_in[39];
  const float* dv_b1 = (const float*)d_in[40];
  const float* dv_w2 = (const float*)d_in[41];
  const float* dv_b2 = (const float*)d_in[42];

  // Output sections (reference return order)
  float* out = (float*)d_out;
  float* out_mean = out;                      // 256*600*32
  float* out_var  = out_mean + 4915200;       // 256*600*32
  float* o_ppm    = out_var + 4915200;        // 256*4*60
  float* o_prpm   = o_ppm + 61440;            // 256*3*60
  float* o_ptm    = o_prpm + 46080;           // 256*40*60
  float* o_prtm   = o_ptm + 614400;           // 256*36*60
  float* o_aa     = o_prtm + 552960;          // 256*3*60

  // Workspace carve
  char* ws = (char*)d_ws;
  size_t off = 0;
  auto ALLOC = [&](size_t bytes) -> char* {
    char* p = ws + off;
    off += (bytes + 255) & ~(size_t)255;
    return p;
  };
  f16* aoeW1s = (f16*)ALLOC(64 * 256 * 2);
  f16* aoeWms = (f16*)ALLOC(16384 * 2);
  f16* aoeWvs = (f16*)ALLOC(16384 * 2);
  f16* aeW1s  = (f16*)ALLOC(32 * 256 * 2);
  f16* aeWms  = (f16*)ALLOC(16384 * 2);
  f16* aeWvs  = (f16*)ALLOC(16384 * 2);
  f16* oeW1s  = (f16*)ALLOC(64 * 256 * 2);
  f16* oeWms  = (f16*)ALLOC(16384 * 2);
  f16* oeWvs  = (f16*)ALLOC(16384 * 2);
  f16* dmW1s  = (f16*)ALLOC(64 * 256 * 2);
  f16* dmW2s  = (f16*)ALLOC(16384 * 2);
  f16* dvW1s  = (f16*)ALLOC(96 * 256 * 2);
  f16* dvW2s  = (f16*)ALLOC(16384 * 2);
  float* encM = (float*)ALLOC((size_t)ROWS * 64 * 4);
  float* encV = (float*)ALLOC((size_t)ROWS * 64 * 4);
  float* S1   = (float*)ALLOC((size_t)Bsz * 40 * 30 * 4);
  float* S2   = (float*)ALLOC((size_t)Bsz * 40 * 30 * 4);
  float* SA   = (float*)ALLOC((size_t)Bsz * 40 * 60 * 4);
  float* SB   = (float*)ALLOC((size_t)Bsz * 40 * 60 * 4);
  float* bseq = (float*)ALLOC((size_t)ROWS * 60 * 4);
  float* psm  = (float*)ALLOC((size_t)ROWS * 60 * 4);
  float* psc  = (float*)ALLOC((size_t)ROWS * 90 * 4);

  auto convW1 = [&](const float* s, int sr, f16* d, int kpad) {
    int n = kpad * 256;
    conv_w1_swz_kernel<<<(n + 255) / 256, 256, 0, stream>>>(s, sr, d, kpad);
  };
  auto convHead = [&](const float* s, int sc, f16* d) {
    conv_head_swz_kernel<<<64, 256, 0, stream>>>(s, sc, d);
  };
  convW1(aoe_w1, 33, aoeW1s, 64);
  convHead(aoe_wm, 30, aoeWms);
  convHead(aoe_wv, 30, aoeWvs);
  convW1(ae_w1, 9, aeW1s, 32);
  convHead(ae_wm, 60, aeWms);
  convHead(ae_wv, 60, aeWvs);
  convW1(oe_w1, 32, oeW1s, 64);
  convHead(oe_wm, 30, oeWms);
  convHead(oe_wv, 30, oeWvs);
  convW1(dm_w1, 60, dmW1s, 64);
  convHead(dm_w2, 32, dmW2s);
  convW1(dv_w1, 90, dvW1s, 96);
  convHead(dv_w2, 32, dvW2s);

  const int gENC = ROWS / 64;  // 2400

  // ---- Level 1 ----
  enc_wmma_kernel<<<gENC, 256, 0, stream>>>(obs, 32, 150, aoeW1s, 64, aoe_b1,
      aoeWms, aoe_bm, aoeWvs, aoe_bv, encM, encV, 64, 30, ROWS, 0);
  reduce_obs_kernel<<<(Bsz * 4 * 30 + 255) / 256, 256, 0, stream>>>(encM, encV, valid, 150, 4, S1, S2);
  enc_wmma_kernel<<<gENC, 256, 0, stream>>>(act, 8, 150, aeW1s, 32, ae_b1,
      aeWms, ae_bm, aeWvs, ae_bv, encM, encV, 64, 60, ROWS, 0);
  reduce_act_kernel<<<(Bsz * 4 * 60 + 255) / 256, 256, 0, stream>>>(encM, encV, 150, 4, SA, SB);
  level_rec_kernel<<<Bsz, 64, 0, stream>>>(S1, S2, SA, SB, pp_wc, pp_bc, pp_tm, pp_ln,
                                           o_ppm, o_prpm, o_aa, 4, 4, 3);

  // ---- Level 2 ----
  enc_wmma_kernel<<<gENC, 256, 0, stream>>>(obs, 32, 15, aoeW1s, 64, aoe_b1,
      aoeWms, aoe_bm, aoeWvs, aoe_bv, encM, encV, 64, 30, ROWS, 0);
  reduce_obs_kernel<<<(Bsz * 40 * 30 + 255) / 256, 256, 0, stream>>>(encM, encV, valid, 15, 40, S1, S2);
  enc_wmma_kernel<<<gENC, 256, 0, stream>>>(act, 8, 15, aeW1s, 32, ae_b1,
      aeWms, ae_bm, aeWvs, ae_bv, encM, encV, 64, 60, ROWS, 0);
  reduce_act_kernel<<<(Bsz * 40 * 60 + 255) / 256, 256, 0, stream>>>(encM, encV, 15, 40, SA, SB);
  level_rec_kernel<<<Bsz, 64, 0, stream>>>(S1, S2, SA, SB, tp_wc, tp_bc, tp_tm, tp_ln,
                                           o_ptm, o_prtm, nullptr, 40, 10, 36);

  // ---- Level 3 ----
  enc_wmma_kernel<<<gENC, 256, 0, stream>>>(obs, 32, 0, oeW1s, 64, oe_b1,
      oeWms, oe_bm, oeWvs, oe_bv, encM, encV, 64, 30, ROWS, 0);
  bseq_kernel<<<(ROWS * 60 + 255) / 256, 256, 0, stream>>>(act, o_ptm, sp_wctrl, sp_wtask, bseq);
  scan_kernel<<<Bsz, 64, 0, stream>>>(encM, encV, valid, bseq,
                                      sp_wc, sp_bc, sp_tm, sp_ln, psm, psc);

  // ---- Decoders ----
  enc_wmma_kernel<<<gENC, 256, 0, stream>>>(psm, 60, 0, dmW1s, 64, dm_b1,
      dmW2s, dm_b2, nullptr, nullptr, out_mean, nullptr, 32, 32, ROWS, 0);
  enc_wmma_kernel<<<gENC, 256, 0, stream>>>(psc, 90, 0, dvW1s, 96, dv_b1,
      dvW2s, dv_b2, nullptr, nullptr, out_var, nullptr, 32, 32, ROWS, 1);
}
